// MultiHeadAttention_32993938768231
// MI455X (gfx1250) — matrix operand order repaired and checked by reading, not compiled, not run on hardware
//
#include <hip/hip_runtime.h>

#define HID 1024
#define SEQ 2048
#define BAT 2
#define NH  16
#define HD  64
#define MTOT (BAT*SEQ)   // 4096 rows for the projection GEMMs
#define LSTR 2052        // padded logits row stride (floats): conflict-free score writes

// -------- CDNA5 async global<->LDS (guarded; per-compile-pass safe) --------
#if defined(__has_builtin)
#  if __has_builtin(__builtin_amdgcn_global_load_async_to_lds_b128) && \
      __has_builtin(__builtin_amdgcn_global_store_async_from_lds_b128)
#    define HAVE_ASYNC_LDS 1
#  else
#    define HAVE_ASYNC_LDS 0
#  endif
#else
#  define HAVE_ASYNC_LDS 0
#endif

typedef int v4i __attribute__((ext_vector_type(4)));
typedef __attribute__((address_space(1))) v4i GV4i;  // global b128 chunk
typedef __attribute__((address_space(3))) v4i LV4i;  // LDS b128 chunk

__device__ __forceinline__ void wait_asynccnt0() {
#if defined(__has_builtin)
#  if __has_builtin(__builtin_amdgcn_s_wait_asynccnt)
  __builtin_amdgcn_s_wait_asynccnt(0);
#  else
  asm volatile("s_wait_asynccnt 0" ::: "memory");
#  endif
#else
  asm volatile("s_wait_asynccnt 0" ::: "memory");
#endif
}

typedef __attribute__((ext_vector_type(16))) __bf16 bf16x16;
typedef __attribute__((ext_vector_type(8)))  float  f32x8;

union Frag {
  bf16x16 v;
  unsigned int u[8];
  unsigned short s[16];
};

// ---- fp32 -> bf16 (round-to-nearest-even on bits; no __bf16 arithmetic) ----
__device__ __forceinline__ unsigned short f2bf(float f) {
  unsigned u = __float_as_uint(f);
  u += 0x7FFFu + ((u >> 16) & 1u);
  return (unsigned short)(u >> 16);
}
__device__ __forceinline__ unsigned int f2bf2(float lo, float hi) {
  return (unsigned)f2bf(lo) | ((unsigned)f2bf(hi) << 16);
}

// A-matrix 16x32 bf16 VGPR layout: VGPR v holds a K pair; v<4 -> K=half*8+2v,
// v>=4 -> K=16+half*8+2(v-4)  (ISA 7.12.2)
__device__ __forceinline__ int a_kk(int v, int half) {
  return (v < 4) ? (half * 8 + v * 2) : (16 + half * 8 + (v - 4) * 2);
}

__device__ __forceinline__ f32x8 wmma_bf16(const Frag& a, const Frag& b, f32x8 c) {
  return __builtin_amdgcn_wmma_f32_16x16x32_bf16(false, a.v, false, b.v,
                                                 (short)0, c, false, false);
}

// =====================================================================
// Kernel 1: Q/K/V projections.  Y = X*W + b, bf16 head-split.
//   z==0 -> Q  [bh][s][d]
//   z==1 -> K  [bh][s][d]
//   z==2 -> V^T[bh][d][s]  (transposed so ctx-GEMM B-frags are contiguous)
// block = 128 (4 waves), tile 64(M) x 64(N), K-steps of 32.
// =====================================================================
__global__ void __launch_bounds__(128)
qkv_proj_kernel(const float* __restrict__ q, const float* __restrict__ k,
                const float* __restrict__ v,
                const float* __restrict__ Wq, const float* __restrict__ bq,
                const float* __restrict__ Wk, const float* __restrict__ bk,
                const float* __restrict__ Wv, const float* __restrict__ bv,
                unsigned short* __restrict__ Qo, unsigned short* __restrict__ Ko,
                unsigned short* __restrict__ Vto)
{
  __shared__ __align__(16) unsigned short Abuf[64 * 40]; // [m][k], stride 40
  __shared__ __align__(16) unsigned short Bbuf[64 * 40]; // [n][k], stride 40

  const int z = blockIdx.z;
  const float* X    = (z == 0) ? q  : (z == 1) ? k  : v;
  const float* W    = (z == 0) ? Wq : (z == 1) ? Wk : Wv;
  const float* bias = (z == 0) ? bq : (z == 1) ? bk : bv;
  unsigned short* Y = (z == 0) ? Qo : (z == 1) ? Ko : Vto;

  const int m0 = blockIdx.x * 64;
  const int n0 = blockIdx.y * 64;
  const int tid  = threadIdx.x;
  const int wave = tid >> 5, lane = tid & 31;
  const int lm = lane & 15, half = lane >> 4;
  const int wm = wave >> 1, wn = wave & 1;

  f32x8 acc[2][2] = {};

  for (int kb = 0; kb < HID / 32; ++kb) {
    const int k0 = kb * 32;
    // A tile 64x32: float4 global loads, pack 2x bf16-pairs, b64 LDS stores
    for (int i = 0; i < 4; ++i) {
      int pid = tid + i * 128;                 // 512 quads
      int row = pid >> 3, kq = pid & 7;
      float4 f = *(const float4*)(X + (size_t)(m0 + row) * HID + k0 + kq * 4);
      uint2 p2; p2.x = f2bf2(f.x, f.y); p2.y = f2bf2(f.z, f.w);
      *(uint2*)&Abuf[row * 40 + kq * 4] = p2;
    }
    // B tile 32(k)x64(n): float4 loads along n, store transposed [n][k]
    for (int i = 0; i < 4; ++i) {
      int pid = tid + i * 128;                 // 512 quads
      int kr = pid >> 4, nq = pid & 15;
      float4 f = *(const float4*)(W + (size_t)(k0 + kr) * HID + n0 + nq * 4);
      Bbuf[(nq * 4 + 0) * 40 + kr] = f2bf(f.x);
      Bbuf[(nq * 4 + 1) * 40 + kr] = f2bf(f.y);
      Bbuf[(nq * 4 + 2) * 40 + kr] = f2bf(f.z);
      Bbuf[(nq * 4 + 3) * 40 + kr] = f2bf(f.w);
    }
    __syncthreads();

    Frag a[2], b[2];
    for (int mt = 0; mt < 2; ++mt)
      for (int vv = 0; vv < 8; ++vv)
        a[mt].u[vv] = *(const unsigned int*)
            &Abuf[(wm * 32 + mt * 16 + lm) * 40 + a_kk(vv, half)];
    for (int nt = 0; nt < 2; ++nt)
      for (int vv = 0; vv < 8; ++vv)
        b[nt].u[vv] = *(const unsigned int*)
            &Bbuf[(wn * 32 + nt * 16 + lm) * 40 + a_kk(vv, half)];

    for (int mt = 0; mt < 2; ++mt)
      for (int nt = 0; nt < 2; ++nt)
        acc[mt][nt] = wmma_bf16(a[mt], b[nt], acc[mt][nt]);
    __syncthreads();
  }

  // epilogue: +bias, bf16 head-split scatter
  for (int mt = 0; mt < 2; ++mt)
    for (int nt = 0; nt < 2; ++nt)
      for (int r = 0; r < 8; ++r) {
        int gm = m0 + wm * 32 + mt * 16 + r + half * 8;
        int gn = n0 + wn * 32 + nt * 16 + lm;
        float val = acc[mt][nt][r] + bias[gn];
        int bidx = gm >> 11, s = gm & (SEQ - 1);
        int h = gn >> 6, d = gn & (HD - 1);
        size_t idx;
        if (z < 2) idx = (((size_t)(bidx * NH + h)) * SEQ + s) * HD + d;
        else       idx = (((size_t)(bidx * NH + h)) * HD + d) * SEQ + s;
        Y[idx] = f2bf(val);
      }
}

// =====================================================================
// Kernel 2: fused scores -> softmax -> attn write -> ctx, per (bh, 16-row strip)
// block = 256 (8 waves). Dynamic LDS: logits[16][LSTR] f32 + sctx[16][64] f32.
// attn crosses HBM exactly once; store issued async-from-LDS when available.
// =====================================================================
__global__ void __launch_bounds__(256)
attn_kernel(const unsigned short* __restrict__ Q, const unsigned short* __restrict__ K,
            const unsigned short* __restrict__ Vt, float* __restrict__ out,
            unsigned short* __restrict__ ctx)
{
  extern __shared__ float dls[];
  float* logits = dls;                 // 16 * LSTR fp32
  float* sctx   = dls + 16 * LSTR;     // 16 * 64 fp32

  const int s0 = blockIdx.x * 16;
  const int bh = blockIdx.y;           // b*16 + h
  const int tid  = threadIdx.x;
  const int wave = tid >> 5, lane = tid & 31;
  const int lm = lane & 15, half = lane >> 4;

  const unsigned short* Qh = Q  + (size_t)bh * SEQ * HD;
  const unsigned short* Kh = K  + (size_t)bh * SEQ * HD;
  const unsigned short* Vh = Vt + (size_t)bh * HD * SEQ;   // [d][t]
  float* attnp = out + (size_t)MTOT * HID;                 // attn after main out

  // Q strip A-fragments (rows s0..s0+15), 2 K-steps over D=64
  Frag aq[2];
  for (int ks = 0; ks < 2; ++ks)
    for (int vv = 0; vv < 8; ++vv)
      aq[ks].u[vv] = *(const unsigned int*)
          &Qh[(size_t)(s0 + lm) * HD + ks * 32 + a_kk(vv, half)];

  // ---- scores: each wave covers 256 columns (16 tiles) ----
  for (int nt = 0; nt < 16; ++nt) {
    int n0 = wave * 256 + nt * 16;
    f32x8 acc = {};
    for (int ks = 0; ks < 2; ++ks) {
      Frag b;  // B = K^T: lane n=t, k=d pairs contiguous in Kh[t][d]
      for (int vv = 0; vv < 8; ++vv)
        b.u[vv] = *(const unsigned int*)
            &Kh[(size_t)(n0 + lm) * HD + ks * 32 + a_kk(vv, half)];
      acc = wmma_bf16(aq[ks], b, acc);
    }
    for (int r = 0; r < 8; ++r)
      logits[(r + half * 8) * LSTR + n0 + lm] = acc[r] * 0.125f; // 1/sqrt(64)
  }
  for (int i = 0; i < 4; ++i) sctx[tid + i * 256] = 0.0f;
  __syncthreads();

  // ---- softmax: wave w owns rows 2w, 2w+1 (float4-vectorized passes) ----
  for (int rr = 0; rr < 2; ++rr) {
    int row = wave * 2 + rr;
    float* L = logits + row * LSTR;
    float mx = -3.4e38f;
    for (int c = lane * 4; c < SEQ; c += 128) {
      float4 t = *(float4*)&L[c];
      mx = fmaxf(mx, fmaxf(fmaxf(t.x, t.y), fmaxf(t.z, t.w)));
    }
    for (int off = 16; off; off >>= 1) mx = fmaxf(mx, __shfl_xor(mx, off, 32));
    float sum = 0.f;
    for (int c = lane * 4; c < SEQ; c += 128) {
      float4 t = *(float4*)&L[c];
      t.x = __expf(t.x - mx); t.y = __expf(t.y - mx);
      t.z = __expf(t.z - mx); t.w = __expf(t.w - mx);
      *(float4*)&L[c] = t;
      sum += (t.x + t.y) + (t.z + t.w);
    }
    for (int off = 16; off; off >>= 1) sum += __shfl_xor(sum, off, 32);
    float inv = 1.0f / sum;
    size_t base = ((size_t)bh * SEQ + (s0 + row)) * SEQ;
    for (int c = lane * 4; c < SEQ; c += 128) {
      float4 t = *(float4*)&L[c];
      t.x *= inv; t.y *= inv; t.z *= inv; t.w *= inv;
      *(float4*)&L[c] = t;                 // LDS now holds normalized probs
#if !HAVE_ASYNC_LDS
      *(float4*)&attnp[base + c] = t;      // fallback: direct b128 store
#endif
    }
    (void)base;
  }
  __syncthreads();

#if HAVE_ASYNC_LDS
  // ---- attn output: async LDS -> global b128, overlapped with PV WMMAs ----
  for (int rr = 0; rr < 2; ++rr) {
    int row = wave * 2 + rr;
    size_t base = ((size_t)bh * SEQ + (s0 + row)) * SEQ;
    for (int i = 0; i < 16; ++i) {
      int off = (i * 32 + lane) * 4;       // 16B per lane per issue
      __builtin_amdgcn_global_store_async_from_lds_b128(
          (GV4i*)(attnp + base + off),
          (LV4i*)&logits[row * LSTR + off], 0, 0);
    }
  }
#endif

  // ---- ctx = P * V : K-dim (t) split across 8 waves, 8 steps of 32 each ----
  f32x8 accc[4] = {};
  for (int ks = 0; ks < 8; ++ks) {
    int t0 = wave * 256 + ks * 32;
    Frag a;  // A = normalized probs from LDS, converted to bf16
    for (int vv = 0; vv < 8; ++vv) {
      int kk = a_kk(vv, half);
      a.u[vv] = f2bf2(logits[lm * LSTR + t0 + kk],
                      logits[lm * LSTR + t0 + kk + 1]);
    }
    for (int nt = 0; nt < 4; ++nt) {
      Frag b;  // B = V: lane n=d, k=t pairs contiguous in Vh[d][t]
      for (int vv = 0; vv < 8; ++vv)
        b.u[vv] = *(const unsigned int*)
            &Vh[(size_t)(nt * 16 + lm) * SEQ + t0 + a_kk(vv, half)];
      accc[nt] = wmma_bf16(a, b, accc[nt]);
    }
  }
  // reduce partial ctx across waves via LDS float atomics (ds_add_f32)
  for (int nt = 0; nt < 4; ++nt)
    for (int r = 0; r < 8; ++r)
      atomicAdd(&sctx[(r + half * 8) * HD + nt * 16 + lm], accc[nt][r]);
  __syncthreads();

  // ---- write ctx bf16 [B,S,HID] ----
  int b = bh >> 4, h = bh & 15;
  for (int i = 0; i < 4; ++i) {
    int idx = tid + i * 256;
    int r = idx >> 6, d = idx & 63;
    float val = sctx[idx];
    size_t o = ((size_t)(b * SEQ + s0 + r)) * HID + h * HD + d;
    ctx[o] = f2bf(val);
  }
}

// =====================================================================
// Kernel 3: out = ctx(bf16) * Wo + bo -> fp32 d_out[0 .. 4M)
// A tile is already bf16 -> async global->LDS copy when available.
// =====================================================================
__global__ void __launch_bounds__(128)
out_proj_kernel(const unsigned short* __restrict__ ctxb, const float* __restrict__ Wo,
                const float* __restrict__ bo, float* __restrict__ out)
{
  __shared__ __align__(16) unsigned short Abuf[64 * 40];
  __shared__ __align__(16) unsigned short Bbuf[64 * 40];

  const int m0 = blockIdx.x * 64;
  const int n0 = blockIdx.y * 64;
  const int tid  = threadIdx.x;
  const int wave = tid >> 5, lane = tid & 31;
  const int lm = lane & 15, half = lane >> 4;
  const int wm = wave >> 1, wn = wave & 1;

  f32x8 acc[2][2] = {};

  for (int kb = 0; kb < HID / 32; ++kb) {
    const int k0 = kb * 32;
#if HAVE_ASYNC_LDS
    // A tile 64 rows x 64B, as 256 16B chunks via async global->LDS
    for (int i = 0; i < 2; ++i) {
      int cid = tid + i * 128;
      int row = cid >> 2, ch = cid & 3;
      __builtin_amdgcn_global_load_async_to_lds_b128(
          (GV4i*)(ctxb + (size_t)(m0 + row) * HID + k0 + ch * 8),
          (LV4i*)&Abuf[row * 40 + ch * 8], 0, 0);
    }
#else
    for (int i = 0; i < 2; ++i) {
      int cid = tid + i * 128;
      int row = cid >> 2, ch = cid & 3;
      *(uint4*)&Abuf[row * 40 + ch * 8] =
          *(const uint4*)&ctxb[(size_t)(m0 + row) * HID + k0 + ch * 8];
    }
#endif
    for (int i = 0; i < 4; ++i) {
      int pid = tid + i * 128;
      int kr = pid >> 4, nq = pid & 15;
      float4 f = *(const float4*)(Wo + (size_t)(k0 + kr) * HID + n0 + nq * 4);
      Bbuf[(nq * 4 + 0) * 40 + kr] = f2bf(f.x);
      Bbuf[(nq * 4 + 1) * 40 + kr] = f2bf(f.y);
      Bbuf[(nq * 4 + 2) * 40 + kr] = f2bf(f.z);
      Bbuf[(nq * 4 + 3) * 40 + kr] = f2bf(f.w);
    }
#if HAVE_ASYNC_LDS
    wait_asynccnt0();
#endif
    __syncthreads();

    Frag a[2], b[2];
    for (int mt = 0; mt < 2; ++mt)
      for (int vv = 0; vv < 8; ++vv)
        a[mt].u[vv] = *(const unsigned int*)
            &Abuf[(wm * 32 + mt * 16 + lm) * 40 + a_kk(vv, half)];
    for (int nt = 0; nt < 2; ++nt)
      for (int vv = 0; vv < 8; ++vv)
        b[nt].u[vv] = *(const unsigned int*)
            &Bbuf[(wn * 32 + nt * 16 + lm) * 40 + a_kk(vv, half)];

    for (int mt = 0; mt < 2; ++mt)
      for (int nt = 0; nt < 2; ++nt)
        acc[mt][nt] = wmma_bf16(a[mt], b[nt], acc[mt][nt]);
    __syncthreads();
  }

  for (int mt = 0; mt < 2; ++mt)
    for (int nt = 0; nt < 2; ++nt)
      for (int r = 0; r < 8; ++r) {
        int gm = m0 + wm * 32 + mt * 16 + r + half * 8;
        int gn = n0 + wn * 32 + nt * 16 + lm;
        out[(size_t)gm * HID + gn] = acc[mt][nt][r] + bo[gn];
      }
}

// =====================================================================
extern "C" void kernel_launch(void* const* d_in, const int* in_sizes, int n_in,
                              void* d_out, int out_size, void* d_ws, size_t ws_size,
                              hipStream_t stream) {
  (void)in_sizes; (void)n_in; (void)out_size; (void)ws_size;
  const float* q  = (const float*)d_in[0];
  const float* k  = (const float*)d_in[1];
  const float* v  = (const float*)d_in[2];
  const float* Wq = (const float*)d_in[3];
  const float* bq = (const float*)d_in[4];
  const float* Wk = (const float*)d_in[5];
  const float* bk = (const float*)d_in[6];
  const float* Wv = (const float*)d_in[7];
  const float* bv = (const float*)d_in[8];
  const float* Wo = (const float*)d_in[9];
  const float* bo = (const float*)d_in[10];
  float* out = (float*)d_out;

  // workspace: Q, K, V^T, ctx — all bf16 [4096 x 1024] = 8MB each (32MB total)
  unsigned short* Qb = (unsigned short*)d_ws;
  unsigned short* Kb = Qb + (size_t)MTOT * HID;
  unsigned short* Vt = Kb + (size_t)MTOT * HID;
  unsigned short* Ct = Vt + (size_t)MTOT * HID;

  qkv_proj_kernel<<<dim3(MTOT / 64, HID / 64, 3), 128, 0, stream>>>(
      q, k, v, Wq, bq, Wk, bk, Wv, bv, Qb, Kb, Vt);

  size_t shl = (size_t)(16 * LSTR + 16 * HD) * sizeof(float); // ~132KB dynamic LDS
  attn_kernel<<<dim3(SEQ / 16, BAT * NH), 256, shl, stream>>>(Qb, Kb, Vt, out, Ct);

  out_proj_kernel<<<dim3(MTOT / 64, HID / 64), 128, 0, stream>>>(Ct, Wo, bo, out);
}
